// AxialSelfAttentionModule_2482491097688
// MI455X (gfx1250) — matrix operand order repaired and checked by reading, not compiled, not run on hardware
//
#include <hip/hip_runtime.h>

// ---------------------------------------------------------------------------
// AxialSelfAttention forward for MI455X (gfx1250, wave32).
//
// Math plan (forward-exact simplifications):
//  * procrustes rotations are orthogonal and applied to BOTH q and k ->
//    q.k invariant -> R9_* dead.
//  * pa bias is constant along the softmax axis -> softmax invariant ->
//    paw/pab dead.
//  * QKV is pointwise -> identical across the three axial passes -> once.
//
// HW plan:
//  * All matmuls (3^3 circular conv as 27-tap GEMM, 1x1 convs, QKV, QK^T, PV,
//    projection) via v_wmma_f32_16x16x32_f16, f32 accumulation.
//  * Activations kept position-major f16 [P][192] in global memory so every
//    GEMM tile is a clean 2D Tensor-Data-Mover descriptor (dim0 = 32
//    contiguous k f16, dim1 = rows, stride 192). TDM lands tiles in LDS
//    K-innermost -> every WMMA fragment gather is contiguous ds_load_b128.
//  * One wave per block issues tensor_load_to_lds, waits TENSORcnt==0, and a
//    workgroup barrier releases the consumers.
//  * Circular w-shift for the conv = <=2 TDM descriptors (split at the wrap);
//    d/h shifts only move the row base. Conv weights pre-transposed [tap][O][C].
// ---------------------------------------------------------------------------

typedef __attribute__((ext_vector_type(16))) _Float16     v16h;
typedef __attribute__((ext_vector_type(8)))  float        v8f;
typedef __attribute__((ext_vector_type(4)))  unsigned int u32x4;
typedef __attribute__((ext_vector_type(8)))  int          i32x8;
typedef __attribute__((ext_vector_type(4)))  int          i32x4;

#define WMMA(a, b, c) \
    __builtin_amdgcn_wmma_f32_16x16x32_f16(false, (a), false, (b), (short)0, (c), false, false)

#if defined(__has_builtin)
#if __has_builtin(__builtin_amdgcn_tensor_load_to_lds)
#define HAVE_TDM 1
#endif
#endif

__device__ __forceinline__ float gelu_exact(float z) {
    return 0.5f * z * (1.0f + erff(z * 0.70710678118654752f));
}

// 2D tile global->LDS: tile0 = 32 contiguous f16, tile1 rows, row stride
// `stride_elems` f16.  TDM D# per CDNA5 ISA ch.8 (group0/group1 bitfields).
__device__ __forceinline__ void tdm_load_2d(_Float16* ldsp, const void* gptr,
                                            unsigned tile1, unsigned stride_elems,
                                            int lane)
{
#ifdef HAVE_TDM
    const unsigned long long ga = (unsigned long long)(size_t)gptr;
    const unsigned lds = (unsigned)(size_t)(void*)ldsp;
    u32x4 g0;
    g0[0] = 1u;                                              // count=1, user D#
    g0[1] = lds;                                             // lds_addr
    g0[2] = (unsigned)(ga & 0xffffffffull);                  // global_addr lo
    g0[3] = (unsigned)((ga >> 32) & 0x01ffffffull) | (2u << 30); // hi | type=2
    const unsigned td0 = 0x40000000u, td1 = 0x40000000u;     // huge dims: no OOB
    i32x8 g1;
    g1[0] = (int)(1u << 16);                                 // data_size = 2B
    g1[1] = (int)((td0 & 0xffffu) << 16);                    // tensor_dim0 lo16
    g1[2] = (int)((td0 >> 16) | ((td1 & 0xffffu) << 16));    // dim0 hi | dim1 lo
    g1[3] = (int)((td1 >> 16) | (32u << 16));                // dim1 hi | tile_dim0=32
    g1[4] = (int)(tile1 & 0xffffu);                          // tile_dim1, tile_dim2=0
    g1[5] = (int)stride_elems;                               // dim0_stride lo32
    g1[6] = 0;                                               // dim0_stride hi16
    g1[7] = 0;
    i32x4 z4 = {0, 0, 0, 0};
    z4[0] = 0x40000000;                                      // tensor_dim2 (unused)
#if __clang_major__ >= 23
    i32x8 z8 = {0, 0, 0, 0, 0, 0, 0, 0};
    __builtin_amdgcn_tensor_load_to_lds(g0, g1, z4, z4, z8, 0);
#else
    __builtin_amdgcn_tensor_load_to_lds(g0, g1, z4, z4, 0);
#endif
#else
    const _Float16* g = (const _Float16*)gptr;
    for (unsigned i = lane; i < 32u * tile1; i += 32u) {
        const unsigned y = i >> 5, x0 = i & 31;
        ldsp[y * 32 + x0] = g[(size_t)y * stride_elems + x0];
    }
#endif
}

__device__ __forceinline__ void tdm_wait_all() {
#ifdef HAVE_TDM
    __builtin_amdgcn_s_wait_tensorcnt(0);
#endif
}

// ---------------------------------------------------------------------------
// gemm192: Y[p,o] = sum_{k,tap} A[p_shift_tap][k] * Wt[tap][o][k] + bias[o]
//   A   : position-major f16 [P][192]
//   Wt  : f16 [taps][O][192]
//   out : store_cm ? f32 channel-major (B,O,32768) (final) : f16 PM [P][O]
//   block = 128 thr = 4 waves; M-tile 32 positions (one w-row), N-tile 64.
// ---------------------------------------------------------------------------
__global__ __launch_bounds__(128)
void gemm192_kernel(const _Float16* __restrict__ A,
                    const _Float16* __restrict__ Wt,
                    const float* __restrict__ bias,
                    _Float16* __restrict__ Ypm,
                    float* __restrict__ Ycm,
                    int O, int taps, int store_cm)
{
    __shared__ _Float16 As[32 * 32];   // [m][k] K-innermost
    __shared__ _Float16 Bs[64 * 32];   // [n][k] K-innermost

    const int t    = threadIdx.x;
    const int wave = t >> 5;
    const int lane = t & 31;
    const int nl   = lane & 15;
    const bool hi  = lane >= 16;

    const int rt = blockIdx.x;                 // 2048 w-row tiles
    const int b  = rt >> 10;
    const int d  = (rt >> 5) & 31;
    const int h  = rt & 31;
    const int n0 = blockIdx.y << 6;

    v8f acc0 = {}, acc1 = {};

    for (int tap = 0; tap < taps; ++tap) {
        int sh, dd, hh;
        if (taps == 27) {
            const int td = tap / 9, th = (tap / 3) % 3, tw = tap % 3;
            sh = tw - 1; dd = (d + td + 31) & 31; hh = (h + th + 31) & 31;
        } else { sh = 0; dd = d; hh = h; }
        const int s0   = (sh + 32) & 31;       // first source w of lds row 0
        const int cnt1 = 32 - s0, cnt2 = s0;   // rows before / after the wrap
        const size_t rowbase = ((size_t)b << 15) + ((size_t)dd << 10) + ((size_t)hh << 5);
        const _Float16* Wtap = Wt + ((size_t)tap * O + n0) * 192;

        for (int kc = 0; kc < 192; kc += 32) {
            __syncthreads();                   // previous tile fully consumed
            if (wave == 0) {
                tdm_load_2d(Bs, Wtap + kc, 64, 192, lane);
                tdm_load_2d(As, A + (rowbase + s0) * 192 + kc, cnt1, 192, lane);
                if (cnt2)
                    tdm_load_2d(As + cnt1 * 32, A + rowbase * 192 + kc, cnt2, 192, lane);
                tdm_wait_all();
            }
            __syncthreads();                   // tiles visible to all waves

            const _Float16* Am0 = As + nl * 32;
            const _Float16* Am1 = As + (16 + nl) * 32;
            const _Float16* Bn  = Bs + ((wave << 4) + nl) * 32;
            const int ka0 = hi ? 8 : 0;
            const int kb0 = hi ? 16 : 0;
            v16h a0, a1, bf;
            #pragma unroll
            for (int i = 0; i < 8; ++i) {
                a0[i] = Am0[ka0 + i]; a0[8 + i] = Am0[16 + ka0 + i];
                a1[i] = Am1[ka0 + i]; a1[8 + i] = Am1[16 + ka0 + i];
            }
            #pragma unroll
            for (int i = 0; i < 8; ++i) { bf[i] = Bn[ka0 + i]; bf[8 + i] = Bn[16 + ka0 + i]; }
            acc0 = WMMA(a0, bf, acc0);
            acc1 = WMMA(a1, bf, acc1);
        }
    }

    const int ncol = n0 + (wave << 4) + nl;
    const float bv = bias[ncol];
    const int spb  = (d << 10) + (h << 5);
    #pragma unroll
    for (int r = 0; r < 8; ++r) {
        const int m0 = r + (hi ? 8 : 0);
        const float v0 = acc0[r] + bv;
        const float v1 = acc1[r] + bv;
        if (store_cm) {
            float* dst = Ycm + (((size_t)b * O + ncol) << 15) + spb;
            dst[m0]      = v0;
            dst[m0 + 16] = v1;
        } else {
            const size_t p0 = ((size_t)b << 15) + spb;
            Ypm[(p0 + m0)      * (size_t)O + ncol] = (_Float16)v0;
            Ypm[(p0 + m0 + 16) * (size_t)O + ncol] = (_Float16)v1;
        }
    }
}

// ---------------------------------------------------------------------------
// Converts / elementwise
// ---------------------------------------------------------------------------
__global__ __launch_bounds__(256)
void cvt_lin_kernel(const float* __restrict__ s, _Float16* __restrict__ d, size_t n)
{
    for (size_t i = (size_t)blockIdx.x * 256 + threadIdx.x; i < n;
         i += (size_t)gridDim.x * 256)
        d[i] = (_Float16)s[i];
}

// c1w (O,C,3,3,3) f32 -> [tap][O][C] f16
__global__ __launch_bounds__(256)
void cvt_c1w_kernel(const float* __restrict__ s, _Float16* __restrict__ d)
{
    for (size_t i = (size_t)blockIdx.x * 256 + threadIdx.x; i < 995328u;
         i += (size_t)gridDim.x * 256) {
        const int tap = (int)(i / 36864);
        const int r   = (int)(i % 36864);     // o*192 + c
        d[i] = (_Float16)s[(size_t)r * 27 + tap];
    }
}

// pos_emb (B,C,32768) f32 -> PM f16 [P][192]
__global__ __launch_bounds__(192)
void cvt_pose_kernel(const float* __restrict__ s, _Float16* __restrict__ d)
{
    const int p = blockIdx.x, c = threadIdx.x;
    const int b = p >> 15, sp = p & 32767;
    d[(size_t)p * 192 + c] = (_Float16)s[(((size_t)b * 192 + c) << 15) + sp];
}

// x_mod = x * sigmoid(mraw)  (x CM f32, mraw/xmod PM f16)
__global__ __launch_bounds__(192)
void modmul_pm_kernel(const float* __restrict__ x, const _Float16* __restrict__ mraw,
                      _Float16* __restrict__ xmod)
{
    const int p = blockIdx.x, c = threadIdx.x;
    const int b = p >> 15, sp = p & 32767;
    const float xv = x[(((size_t)b * 192 + c) << 15) + sp];
    const float mr = (float)mraw[(size_t)p * 192 + c];
    xmod[(size_t)p * 192 + c] = (_Float16)(xv * (1.0f / (1.0f + __expf(-mr))));
}

__global__ __launch_bounds__(256)
void zero_kernel(float* __restrict__ p, size_t n)
{
    for (size_t i = (size_t)blockIdx.x * 256 + threadIdx.x; i < n;
         i += (size_t)gridDim.x * 256)
        p[i] = 0.0f;
}

// In-place InstanceNorm (eps=1e-5, biased var) + exact GELU on PM f16 [P][192].
// One block per (b,c); f32 accumulation.
__global__ __launch_bounds__(256)
void inorm_gelu_pm_kernel(_Float16* __restrict__ Z)
{
    const int bc = blockIdx.x;                 // b*192 + c
    const int b = bc / 192, c = bc % 192;
    _Float16* base = Z + (((size_t)b << 15)) * 192 + c;
    float s = 0.0f, s2 = 0.0f;
    for (int i = threadIdx.x; i < 32768; i += 256) {
        const float v = (float)base[(size_t)i * 192];
        s += v; s2 += v * v;
    }
    __shared__ float rs[256], rq[256];
    rs[threadIdx.x] = s; rq[threadIdx.x] = s2;
    __syncthreads();
    for (int off = 128; off > 0; off >>= 1) {
        if (threadIdx.x < off) {
            rs[threadIdx.x] += rs[threadIdx.x + off];
            rq[threadIdx.x] += rq[threadIdx.x + off];
        }
        __syncthreads();
    }
    const float mean = rs[0] * (1.0f / 32768.0f);
    const float var  = rq[0] * (1.0f / 32768.0f) - mean * mean;
    const float rstd = rsqrtf(var + 1e-5f);
    for (int i = threadIdx.x; i < 32768; i += 256) {
        const float z = ((float)base[(size_t)i * 192] - mean) * rstd;
        base[(size_t)i * 192] = (_Float16)gelu_exact(z);
    }
}

// ---------------------------------------------------------------------------
// Axial attention: one wave per (sequence, head).  LDS rows zero-padded to 32
// so every fragment gather is contiguous b128; v stored transposed [j][key].
// ---------------------------------------------------------------------------
__global__ __launch_bounds__(256)
void axial_attn_kernel(const _Float16* __restrict__ qkv,  // PM f16 [P][576]
                       float* __restrict__ outsum,        // PM f32 [P][192]
                       int axis)                          // 0=D, 1=H, 2=W
{
    const int wave = threadIdx.x >> 5;
    const int lane = threadIdx.x & 31;
    const int task = blockIdx.x * 8 + wave;    // 2048 seq * 16 heads
    const int s    = task >> 4;
    const int head = task & 15;
    const int b  = s >> 10;
    const int i1 = (s >> 5) & 31;
    const int i2 = s & 31;

    int base, stride;
    if (axis == 2)      { base = (i1 << 10) + (i2 << 5); stride = 1;    } // W
    else if (axis == 1) { base = (i1 << 10) + i2;        stride = 32;   } // H
    else                { base = (i1 << 5)  + i2;        stride = 1024; } // D

    // per-wave slice: qs[32][32] ks[32][32] vT[16][32] pr[32][32]  (f16)
    __shared__ _Float16 sh[8][3584];
    _Float16* qs = &sh[wave][0];
    _Float16* ks = qs + 1024;
    _Float16* vT = ks + 1024;
    _Float16* pr = vT + 512;

    {   // lane r loads row r of q,k,v (12 f16 each), zero-pad
        const int r = lane;
        const size_t p = ((size_t)b << 15) + base + (size_t)r * stride;
        const _Float16* src = qkv + p * 576 + head * 12;
        const _Float16 z = (_Float16)0.0f;
        #pragma unroll
        for (int j = 0; j < 32; ++j) {
            qs[r * 32 + j] = (j < 12) ? src[j]       : z;
            ks[r * 32 + j] = (j < 12) ? src[192 + j] : z;
        }
        #pragma unroll
        for (int j = 0; j < 16; ++j)
            vT[j * 32 + r] = (j < 12) ? src[384 + j] : z;
    }
    __syncthreads();

    const int nl  = lane & 15;
    const bool hi = lane >= 16;
    const int ka0 = hi ? 8 : 0;
    const int kb0 = hi ? 16 : 0;

    // ---- scores = q k^T (K padded 12->32 with zeros) ----
    v16h a0, a1, bk0, bk1;
    {
        const _Float16* q0 = qs + nl * 32;
        const _Float16* q1 = qs + (16 + nl) * 32;
        const _Float16* k0 = ks + nl * 32;
        const _Float16* k1 = ks + (16 + nl) * 32;
        #pragma unroll
        for (int i = 0; i < 8; ++i) {
            a0[i] = q0[ka0 + i]; a0[8 + i] = q0[16 + ka0 + i];
            a1[i] = q1[ka0 + i]; a1[8 + i] = q1[16 + ka0 + i];
        }
        #pragma unroll
        for (int i = 0; i < 8; ++i) { bk0[i] = k0[ka0 + i]; bk0[8 + i] = k0[16 + ka0 + i]; bk1[i] = k1[ka0 + i]; bk1[8 + i] = k1[16 + ka0 + i]; }
    }
    v8f s00 = {}, s01 = {}, s10 = {}, s11 = {};
    s00 = WMMA(a0, bk0, s00);
    s01 = WMMA(a0, bk1, s01);
    s10 = WMMA(a1, bk0, s10);
    s11 = WMMA(a1, bk1, s11);

    // ---- softmax over keys (columns), scale = 1/sqrt(12) ----
    const float scale = 0.28867513459481287f;
    #pragma unroll
    for (int r = 0; r < 8; ++r) {
        float x0 = s00[r] * scale, x1 = s01[r] * scale;
        float mx = fmaxf(x0, x1);
        for (int off = 1; off < 16; off <<= 1) mx = fmaxf(mx, __shfl_xor(mx, off, 32));
        float e0 = __expf(x0 - mx), e1 = __expf(x1 - mx);
        float sm = e0 + e1;
        for (int off = 1; off < 16; off <<= 1) sm += __shfl_xor(sm, off, 32);
        float inv = 1.0f / sm;
        s00[r] = e0 * inv; s01[r] = e1 * inv;

        x0 = s10[r] * scale; x1 = s11[r] * scale;
        mx = fmaxf(x0, x1);
        for (int off = 1; off < 16; off <<= 1) mx = fmaxf(mx, __shfl_xor(mx, off, 32));
        e0 = __expf(x0 - mx); e1 = __expf(x1 - mx);
        sm = e0 + e1;
        for (int off = 1; off < 16; off <<= 1) sm += __shfl_xor(sm, off, 32);
        inv = 1.0f / sm;
        s10[r] = e0 * inv; s11[r] = e1 * inv;
    }

    // C-layout -> LDS transpose of probs
    #pragma unroll
    for (int r = 0; r < 8; ++r) {
        const int m = (hi ? 8 : 0) + r;
        pr[m * 32 + nl]              = (_Float16)s00[r];
        pr[m * 32 + 16 + nl]         = (_Float16)s01[r];
        pr[(16 + m) * 32 + nl]       = (_Float16)s10[r];
        pr[(16 + m) * 32 + 16 + nl]  = (_Float16)s11[r];
    }
    __syncthreads();

    // ---- out = probs @ v (K=32 exact, N padded 12->16) ----
    v16h pa0, pa1, bv;
    {
        const _Float16* p0 = pr + nl * 32;
        const _Float16* p1 = pr + (16 + nl) * 32;
        const _Float16* vr = vT + nl * 32;
        #pragma unroll
        for (int i = 0; i < 8; ++i) {
            pa0[i] = p0[ka0 + i]; pa0[8 + i] = p0[16 + ka0 + i];
            pa1[i] = p1[ka0 + i]; pa1[8 + i] = p1[16 + ka0 + i];
        }
        #pragma unroll
        for (int i = 0; i < 8; ++i) { bv[i] = vr[ka0 + i]; bv[8 + i] = vr[16 + ka0 + i]; }
    }
    v8f o0 = {}, o1 = {};
    o0 = WMMA(pa0, bv, o0);
    o1 = WMMA(pa1, bv, o1);

    // scatter-add into PM f32 accumulator
    if (nl < 12) {
        const int c = head * 12 + nl;
        float* dst = outsum + (((size_t)b << 15) + base) * 192 + c;
        const size_t step = (size_t)stride * 192;
        #pragma unroll
        for (int r = 0; r < 8; ++r) {
            const int m = (hi ? 8 : 0) + r;
            dst[(size_t)m * step]        += o0[r];
            dst[(size_t)(16 + m) * step] += o1[r];
        }
    }
}

// ---------------------------------------------------------------------------
// Host launcher
// ---------------------------------------------------------------------------
extern "C" void kernel_launch(void* const* d_in, const int* in_sizes, int n_in,
                              void* d_out, int out_size, void* d_ws, size_t ws_size,
                              hipStream_t stream)
{
    const float* x     = (const float*)d_in[0];
    const float* pose  = (const float*)d_in[1];
    const float* Wqkv  = (const float*)d_in[2];
    const float* bqkv  = (const float*)d_in[3];
    const float* c1w   = (const float*)d_in[4];
    const float* c1b   = (const float*)d_in[5];
    const float* c2w   = (const float*)d_in[6];
    const float* c2b   = (const float*)d_in[7];
    const float* m1w   = (const float*)d_in[8];
    const float* m1b   = (const float*)d_in[9];
    const float* m2w   = (const float*)d_in[10];
    const float* m2b   = (const float*)d_in[11];
    // d_in[12..16] = paw, pab, R9_d, R9_h, R9_w: dead in the forward value.
    const float* projw = (const float*)d_in[17];
    const float* projb = (const float*)d_in[18];

    const size_t NP  = 65536;
    const size_t NPC = NP * 192;                 // 12,582,912
    _Float16* pe16 = (_Float16*)d_ws;            // 24 MB
    _Float16* bufA = pe16 + NPC;                 // 24 MB
    _Float16* bufB = bufA + NPC;                 // 24 MB
    _Float16* qkvb = bufB + NPC;                 // 72 MB  [P][576]
    float*    osum = (float*)(qkvb + NP * 576);  // 48 MB  PM f32
    _Float16* wc1t = (_Float16*)(osum + NPC);    // weights arena (~2.5 MB)
    _Float16* wc2  = wc1t + 995328;
    _Float16* wm1  = wc2 + 36864;
    _Float16* wm2  = wm1 + 36864;
    _Float16* wqk  = wm2 + 36864;
    _Float16* wpj  = wqk + 110592;

    // ---- one-time precision/layout conversion of weights + pos_emb ----
    cvt_c1w_kernel<<<512, 256, 0, stream>>>(c1w, wc1t);
    cvt_lin_kernel<<<64, 256, 0, stream>>>(c2w, wc2, 36864);
    cvt_lin_kernel<<<64, 256, 0, stream>>>(m1w, wm1, 36864);
    cvt_lin_kernel<<<64, 256, 0, stream>>>(m2w, wm2, 36864);
    cvt_lin_kernel<<<128, 256, 0, stream>>>(Wqkv, wqk, 110592);
    cvt_lin_kernel<<<64, 256, 0, stream>>>(projw, wpj, 36864);
    cvt_pose_kernel<<<65536, 192, 0, stream>>>(pose, pe16);

    const dim3 g3(2048, 3), g9(2048, 9);

    // lf1 = conv3x3x3_circ(pos_emb) + c1b
    gemm192_kernel<<<g3, 128, 0, stream>>>(pe16, wc1t, c1b, bufA, nullptr, 192, 27, 0);
    // t = gelu(inorm(lf1))
    inorm_gelu_pm_kernel<<<384, 256, 0, stream>>>(bufA);
    // lf = conv1x1(t, c2w) + c2b
    gemm192_kernel<<<g3, 128, 0, stream>>>(bufA, wc2, c2b, bufB, nullptr, 192, 1, 0);
    // u = conv1x1(lf, m1w) + m1b
    gemm192_kernel<<<g3, 128, 0, stream>>>(bufB, wm1, m1b, bufA, nullptr, 192, 1, 0);
    // v2 = gelu(inorm(u))
    inorm_gelu_pm_kernel<<<384, 256, 0, stream>>>(bufA);
    // mraw = conv1x1(v2, m2w) + m2b
    gemm192_kernel<<<g3, 128, 0, stream>>>(bufA, wm2, m2b, bufB, nullptr, 192, 1, 0);
    // x_mod = x * sigmoid(mraw)
    modmul_pm_kernel<<<65536, 192, 0, stream>>>(x, bufB, bufA);
    // qkv = x_mod @ Wqkv^T + bqkv (once)
    gemm192_kernel<<<g9, 128, 0, stream>>>(bufA, wqk, bqkv, qkvb, nullptr, 576, 1, 0);
    // out_sum = sum over the three axial attentions
    zero_kernel<<<2048, 256, 0, stream>>>(osum, NPC);
    for (int axis = 0; axis < 3; ++axis)
        axial_attn_kernel<<<4096, 256, 0, stream>>>(qkvb, osum, axis);
    // y = conv1x1(out_sum, projw) + projb -> d_out (B,C,D,H,W) f32
    cvt_lin_kernel<<<4096, 256, 0, stream>>>(osum, bufA, NPC);
    gemm192_kernel<<<g3, 128, 0, stream>>>(bufA, wpj, projb, nullptr, (float*)d_out, 192, 1, 1);
}
